// QLSTM_65481071407482
// MI455X (gfx1250) — matrix operand order repaired and checked by reading, not compiled, not run on hardware
//
#include <hip/hip_runtime.h>

// ---------------------------------------------------------------------------
// QLSTM on MI455X (gfx1250): persistent per-batch-tile blocks, LDS-resident
// f16 weights, v_wmma_f32_16x16x32_f16 for all three matmul stages per step.
// ---------------------------------------------------------------------------

typedef _Float16 half16 __attribute__((ext_vector_type(16)));
typedef _Float16 half8  __attribute__((ext_vector_type(8)));
typedef float    float8 __attribute__((ext_vector_type(8)));

#define TSTEPS   512
#define BATCH    512
#define DIN      128
#define HDIM     64
#define KC       (DIN + HDIM)   // 192
#define COMBP    200            // padded LDS stride (halves) for K=192 rows
#define QKP      72             // padded LDS stride (halves) for K=64 rows
#define MTILE    16
#define NTHREADS 256            // 8 waves (wave32)

// ---- LDS layout (byte offsets), total ~215 KB of the 320 KB WGP LDS -------
#define SWFC_OFF   0                              // 4*64 rows * 200 h = 102400 B
#define SWQL_OFF   (SWFC_OFF + 4*HDIM*COMBP*2)    // 102400
#define SWQE_OFF   (SWQL_OFF + 4*HDIM*QKP*2)      // 139264
#define SCOMB_OFF  (SWQE_OFF + 4*HDIM*QKP*2)      // 176128 : [16][200] f16
#define SPRE_OFF   (SCOMB_OFF + MTILE*COMBP*2)    // 182528 : [4][16][72] f16
#define SY_OFF     (SPRE_OFF + 4*MTILE*QKP*2)     // 191744 : [4][16][72] f16
#define SQ_OFF     (SY_OFF   + 4*MTILE*QKP*2)     // 200960 : [4][16][64] f32
#define SFCB_OFF   (SQ_OFF   + 4*MTILE*HDIM*4)    // 217344 : 256 f32
#define SQLB_OFF   (SFCB_OFF + 256*4)
#define SQEB_OFF   (SQLB_OFF + 256*4)
#define LDS_BYTES  (SQEB_OFF + 256*4)             // 220416 B

__device__ __forceinline__ float sig_f(float x) {
    return 1.0f / (1.0f + __expf(-x));
}
__device__ __forceinline__ float tanh_f(float x) {
    // overflow-safe tanh on the TRANS pipe
    float e = __expf(-2.0f * fabsf(x));
    float r = (1.0f - e) / (1.0f + e);
    return copysignf(r, x);
}

// A operand (M=16,K=32 f16): ISA layout — lanes 0-15 hold M=lane, K {k0..k0+7,
// k0+16..k0+23}; lanes 16-31 hold K {k0+8..k0+15, k0+24..k0+31}. Two b128 loads.
__device__ __forceinline__ half16 lds_load_a(const _Float16* base, int strideH,
                                             int k0, int lane) {
    const _Float16* p = base + (lane & 15) * strideH + k0 + ((lane & 16) ? 8 : 0);
    half16 r;
    ((uint4*)&r)[0] = *(const uint4*)(p);
    ((uint4*)&r)[1] = *(const uint4*)(p + 16);
    return r;
}

// B operand (K=32,N=16 f16): weights are stored [n][k] (exactly the given
// layout), so B(k,n)=W[n][k]; lanes 0-15 take K {k0..k0+7, k0+16..k0+23} at
// n=ncol+lane, lanes 16-31 take K {k0+8..k0+15, k0+24..k0+31}. Two b128 loads.
__device__ __forceinline__ half16 lds_load_b(const _Float16* base, int strideH,
                                             int ncol, int k0, int lane) {
    const _Float16* p = base + (ncol + (lane & 15)) * strideH + k0 + ((lane & 16) ? 8 : 0);
    half16 r;
    ((uint4*)&r)[0] = *(const uint4*)(p);
    ((uint4*)&r)[1] = *(const uint4*)(p + 16);
    return r;
}

__global__ __launch_bounds__(NTHREADS) void qlstm_main(
    const float* __restrict__ x,      // [T][B][DIN]
    const float* __restrict__ fcW,    // [4][64][192]
    const float* __restrict__ fcb,    // [4][64]
    const float* __restrict__ qlW,    // [4][64][64]
    const float* __restrict__ qlb,    // [4][64]
    const float* __restrict__ qeW,    // [4][64][64]
    const float* __restrict__ qeb,    // [4][64]
    float* __restrict__ out)          // [T][B][64] ++ h[B][64] ++ c[B][64]
{
    extern __shared__ char smem[];
    _Float16* sWfc  = (_Float16*)(smem + SWFC_OFF);
    _Float16* sWql  = (_Float16*)(smem + SWQL_OFF);
    _Float16* sWqe  = (_Float16*)(smem + SWQE_OFF);
    _Float16* sComb = (_Float16*)(smem + SCOMB_OFF);
    _Float16* sPre  = (_Float16*)(smem + SPRE_OFF);
    _Float16* sY    = (_Float16*)(smem + SY_OFF);
    float*    sQ    = (float*)   (smem + SQ_OFF);
    float*    sFcb  = (float*)   (smem + SFCB_OFF);
    float*    sQlb  = (float*)   (smem + SQLB_OFF);
    float*    sQeb  = (float*)   (smem + SQEB_OFF);

    const int tid   = threadIdx.x;
    const int wv    = tid >> 5;
    const int lane  = tid & 31;
    const int bbase = blockIdx.x * MTILE;

    // ---- one-time: convert weights f32 -> f16 into LDS (reused 512 steps) --
    {   // fc: 256 rows of 192, one row per thread, padded stride 200
        const float* src = fcW + (size_t)tid * KC;
        _Float16*    dst = sWfc + (size_t)tid * COMBP;
        #pragma unroll 4
        for (int j = 0; j < KC; j += 4) {
            float4 v = *(const float4*)(src + j);
            dst[j]   = (_Float16)v.x; dst[j+1] = (_Float16)v.y;
            dst[j+2] = (_Float16)v.z; dst[j+3] = (_Float16)v.w;
        }
    }
    {   // ql/qe: 256 rows of 64, padded stride 72
        const float* s0 = qlW + (size_t)tid * HDIM;
        const float* s1 = qeW + (size_t)tid * HDIM;
        _Float16* d0 = sWql + (size_t)tid * QKP;
        _Float16* d1 = sWqe + (size_t)tid * QKP;
        #pragma unroll 4
        for (int j = 0; j < HDIM; j += 4) {
            float4 a = *(const float4*)(s0 + j);
            float4 b = *(const float4*)(s1 + j);
            d0[j] = (_Float16)a.x; d0[j+1] = (_Float16)a.y;
            d0[j+2] = (_Float16)a.z; d0[j+3] = (_Float16)a.w;
            d1[j] = (_Float16)b.x; d1[j+1] = (_Float16)b.y;
            d1[j+2] = (_Float16)b.z; d1[j+3] = (_Float16)b.w;
        }
    }
    sFcb[tid] = fcb[tid];
    sQlb[tid] = qlb[tid];
    sQeb[tid] = qeb[tid];

    // cell/hidden state lives in registers: thread owns (m,n) = (e>>6, e&63)
    float creg[4], hreg[4];
    #pragma unroll
    for (int k = 0; k < 4; ++k) {
        creg[k] = 0.0f; hreg[k] = 0.0f;
        int e = tid + k * NTHREADS;
        sComb[(e >> 6) * COMBP + DIN + (e & 63)] = (_Float16)0.0f;  // h0 = 0
    }
    __syncthreads();

    const int xm  = tid >> 4;          // x-load: row within tile
    const int xd0 = (tid & 15) << 3;   // x-load: 8 columns per thread

    for (int t = 0; t < TSTEPS; ++t) {
        // ---- stage 0: x tile -> sComb[:, 0:128] as f16 ---------------------
        {
            const float* src = x + (((size_t)t * BATCH + bbase + xm) * DIN + xd0);
            float4 f0 = ((const float4*)src)[0];
            float4 f1 = ((const float4*)src)[1];
            half8 h;
            h[0] = (_Float16)f0.x; h[1] = (_Float16)f0.y;
            h[2] = (_Float16)f0.z; h[3] = (_Float16)f0.w;
            h[4] = (_Float16)f1.x; h[5] = (_Float16)f1.y;
            h[6] = (_Float16)f1.z; h[7] = (_Float16)f1.w;
            *(uint4*)(sComb + xm * COMBP + xd0) = *(uint4*)&h;
            if (t + 1 < TSTEPS)   // warm L2/WGP$ for the next step's tile
                __builtin_prefetch(src + (size_t)BATCH * DIN, 0, 3);
        }
        __syncthreads();   // also covers previous step's h writes into sComb

        // ---- stage 1: fc — pre[g] = comb(16x192) x Wfc[g]^T, +bias ---------
        #pragma unroll
        for (int it = 0; it < 2; ++it) {
            int tileId = wv * 2 + it;          // 16 tiles = 4 gates x 4 n-tiles
            int gate = tileId >> 2;
            int ncol = (tileId & 3) << 4;
            const _Float16* wg = sWfc + gate * HDIM * COMBP;
            float8 acc = {0.f,0.f,0.f,0.f,0.f,0.f,0.f,0.f};
            #pragma unroll
            for (int kk = 0; kk < KC / 32; ++kk) {
                half16 a = lds_load_a(sComb, COMBP, kk * 32, lane);
                half16 b = lds_load_b(wg, COMBP, ncol, kk * 32, lane);
                acc = __builtin_amdgcn_wmma_f32_16x16x32_f16(
                          false, a, false, b, (short)0, acc, false, false);
            }
            float bias = sFcb[gate * HDIM + ncol + (lane & 15)];
            int mb = (lane & 16) ? 8 : 0;
            _Float16* pd = sPre + gate * MTILE * QKP + ncol + (lane & 15);
            #pragma unroll
            for (int i = 0; i < 8; ++i)
                pd[(mb + i) * QKP] = (_Float16)(acc[i] + bias);
        }
        __syncthreads();

        // ---- stage 2: ql — y[g] = tanh(pre[g](16x64) x Wql[g]^T + b) -------
        #pragma unroll
        for (int it = 0; it < 2; ++it) {
            int tileId = wv * 2 + it;
            int gate = tileId >> 2;
            int ncol = (tileId & 3) << 4;
            const _Float16* ag = sPre + gate * MTILE * QKP;
            const _Float16* wg = sWql + gate * HDIM * QKP;
            float8 acc = {0.f,0.f,0.f,0.f,0.f,0.f,0.f,0.f};
            #pragma unroll
            for (int kk = 0; kk < 2; ++kk) {
                half16 a = lds_load_a(ag, QKP, kk * 32, lane);
                half16 b = lds_load_b(wg, QKP, ncol, kk * 32, lane);
                acc = __builtin_amdgcn_wmma_f32_16x16x32_f16(
                          false, a, false, b, (short)0, acc, false, false);
            }
            float bias = sQlb[gate * HDIM + ncol + (lane & 15)];
            int mb = (lane & 16) ? 8 : 0;
            _Float16* yd = sY + gate * MTILE * QKP + ncol + (lane & 15);
            #pragma unroll
            for (int i = 0; i < 8; ++i)
                yd[(mb + i) * QKP] = (_Float16)tanh_f(acc[i] + bias);
        }
        __syncthreads();

        // ---- stage 3: qe — q[g] = y[g] * sigmoid(y[g] x Wqe[g]^T + b) ------
        #pragma unroll
        for (int it = 0; it < 2; ++it) {
            int tileId = wv * 2 + it;
            int gate = tileId >> 2;
            int ncol = (tileId & 3) << 4;
            const _Float16* ag = sY + gate * MTILE * QKP;
            const _Float16* wg = sWqe + gate * HDIM * QKP;
            float8 acc = {0.f,0.f,0.f,0.f,0.f,0.f,0.f,0.f};
            #pragma unroll
            for (int kk = 0; kk < 2; ++kk) {
                half16 a = lds_load_a(ag, QKP, kk * 32, lane);
                half16 b = lds_load_b(wg, QKP, ncol, kk * 32, lane);
                acc = __builtin_amdgcn_wmma_f32_16x16x32_f16(
                          false, a, false, b, (short)0, acc, false, false);
            }
            float bias = sQeb[gate * HDIM + ncol + (lane & 15)];
            int mb = (lane & 16) ? 8 : 0;
            int n  = ncol + (lane & 15);
            const _Float16* ys = sY + gate * MTILE * QKP + n;
            float* qd = sQ + gate * MTILE * HDIM + n;
            #pragma unroll
            for (int i = 0; i < 8; ++i) {
                float s  = sig_f(acc[i] + bias);
                float yv = (float)ys[(mb + i) * QKP];
                qd[(mb + i) * HDIM] = yv * s;
            }
        }
        __syncthreads();

        // ---- stage 4: gates + cell update (state stays in registers) -------
        #pragma unroll
        for (int k = 0; k < 4; ++k) {
            int e = tid + k * NTHREADS;          // 1024 elems / 256 threads
            int m = e >> 6, n = e & 63;
            float qf = sQ[(0 * MTILE + m) * HDIM + n];
            float qi = sQ[(1 * MTILE + m) * HDIM + n];
            float qg = sQ[(2 * MTILE + m) * HDIM + n];
            float qo = sQ[(3 * MTILE + m) * HDIM + n];
            float f = sig_f(qf), ii = sig_f(qi), g = tanh_f(qg), o = sig_f(qo);
            float c = f * creg[k] + ii * g;
            float h = o * tanh_f(c);
            creg[k] = c; hreg[k] = h;
            out[((size_t)t * BATCH + bbase + m) * HDIM + n] = h;
            sComb[m * COMBP + DIN + n] = (_Float16)h;   // feed next step
        }
        // no barrier needed here: next iteration's post-x-load barrier covers
        // the sComb h-writes before any wave's fc stage reads them.
    }

    // ---- final (h, c) states -----------------------------------------------
    const size_t TBH = (size_t)TSTEPS * BATCH * HDIM;
    #pragma unroll
    for (int k = 0; k < 4; ++k) {
        int e = tid + k * NTHREADS;
        int m = e >> 6, n = e & 63;
        size_t row = (size_t)(bbase + m) * HDIM + n;
        out[TBH + row] = hreg[k];
        out[TBH + (size_t)BATCH * HDIM + row] = creg[k];
    }
}

extern "C" void kernel_launch(void* const* d_in, const int* in_sizes, int n_in,
                              void* d_out, int out_size, void* d_ws, size_t ws_size,
                              hipStream_t stream) {
    (void)in_sizes; (void)n_in; (void)out_size; (void)d_ws; (void)ws_size;
    const float* x   = (const float*)d_in[0];
    const float* fcW = (const float*)d_in[1];
    const float* fcb = (const float*)d_in[2];
    const float* qlW = (const float*)d_in[3];
    const float* qlb = (const float*)d_in[4];
    const float* qeW = (const float*)d_in[5];
    const float* qeb = (const float*)d_in[6];
    float* out = (float*)d_out;

    // allow >64KB dynamic LDS (host-side attribute set; graph-capture safe)
    (void)hipFuncSetAttribute((const void*)qlstm_main,
                              hipFuncAttributeMaxDynamicSharedMemorySize,
                              LDS_BYTES);

    // 32 persistent blocks: one 16-row batch tile each, all 512 time steps.
    qlstm_main<<<BATCH / MTILE, NTHREADS, LDS_BYTES, stream>>>(
        x, fcW, fcb, qlW, qlb, qeW, qeb, out);
}
